// NhanTuDuyNhanDien_33097017983560
// MI455X (gfx1250) — matrix operand order repaired and checked by reading, not compiled, not run on hardware
//
#include <hip/hip_runtime.h>

typedef __attribute__((ext_vector_type(16))) __bf16 v16bf;
typedef __attribute__((ext_vector_type(8)))  float  v8f;

union BF16x16 { v16bf v; unsigned int u[8]; };

#define HW 50176          // 224*224
#define IMG 224
#define NCH 128

static __device__ __forceinline__ unsigned short f2bf(float f) {
    unsigned int x = __float_as_uint(f);
    return (unsigned short)((x + 0x7FFFu + ((x >> 16) & 1u)) >> 16);
}
static __device__ __forceinline__ unsigned int packf(float a, float b) {
    return (unsigned int)f2bf(a) | ((unsigned int)f2bf(b) << 16);
}
static __device__ __forceinline__ v8f vzero8() {
    v8f z = {0.f,0.f,0.f,0.f,0.f,0.f,0.f,0.f};
    return z;
}
static __device__ __forceinline__ float silu(float v) { return v / (1.f + __expf(-v)); }

// B-fragment loaders -------------------------------------------------------
// channel planes [Kdim][HW], fp32 -> cvt+pack
static __device__ __forceinline__ void loadB_planes(const float* Xb, size_t ncol,
                                                    int kb, BF16x16& bu) {
    #pragma unroll
    for (int d = 0; d < 8; ++d) {
        const float* p0 = Xb + (size_t)(kb + 2 * d) * HW + ncol;
        bu.u[d] = packf(p0[0], p0[HW]);
    }
}
// token-major [HW][Kdim], bf16 -> two aligned b128 loads
template<int Kdim>
static __device__ __forceinline__ void loadB_tok(const unsigned short* Xb, size_t ncol,
                                                 int kb, BF16x16& bu) {
    const uint4* p4 = (const uint4*)(Xb + ncol * (size_t)Kdim + kb);
    uint4 a0 = p4[0], a1 = p4[1];
    bu.u[0] = a0.x; bu.u[1] = a0.y; bu.u[2] = a0.z; bu.u[3] = a0.w;
    bu.u[4] = a1.x; bu.u[5] = a1.y; bu.u[6] = a1.z; bu.u[7] = a1.w;
}

// ---------------------------------------------------------------------------
// WMMA GEMM over spatial positions: Y[m][n] = sum_k W[m][k] * X[k][n].
// Each wave owns a 32-column strip (two 16-col B fragments per k-step) so every
// A fragment pulled from LDS feeds two WMMAs -> half the DS bytes per WMMA.
// XL=0: X = channel planes [Kdim][HW] fp32 (+ prefetch of next k-strip).
// XL=1: X = token-major [HW][Kdim] bf16.
// EPI: 0 = bf16 token-major [HW][Mdim] (b128 stores);
//      1 = BN(eps)+residual fp32 planes; 2 = BN(eps)+SiLU fp32 planes.
// ---------------------------------------------------------------------------
template<int Mdim, int Kdim, typename TIn, int EPI, int XL>
__global__ __launch_bounds__(128)
void gemm_wmma(const float* __restrict__ W, const TIn* __restrict__ X,
               void* __restrict__ Y, const float* __restrict__ gamma,
               const float* __restrict__ beta, float eps,
               const float* __restrict__ res)
{
    constexpr int MT = Mdim / 16;
    constexpr int KS = Kdim / 32;
    __shared__ unsigned int wl[KS * MT * 32 * 8];   // A fragments, lane-swizzled

    const int tid  = threadIdx.x;
    const int lane = tid & 31;
    const int wave = tid >> 5;
    const int b    = blockIdx.y;

    // Pack weights into per-lane A-fragment layout (contiguous k = 16*(lane/16) + s, same order as B).
    for (int i = tid; i < KS * MT * 32 * 8; i += 128) {
        int d  = i & 7;
        int l  = (i >> 3) & 31;
        int tm = i >> 8;
        int mt = tm % MT, ks = tm / MT;
        int row = mt * 16 + (l & 15);
        int kb  = ks * 32 + ((l < 16) ? 0 : 16) + 2 * d;
        const float* wr = W + (size_t)row * Kdim + kb;
        wl[i] = packf(wr[0], wr[1]);
    }
    __syncthreads();

    const size_t ncol0 = ((size_t)blockIdx.x * 4 + wave) * 32 + (lane & 15);
    const size_t ncol1 = ncol0 + 16;
    const TIn* Xb = X + (size_t)b * Kdim * HW;

    v8f acc0[MT], acc1[MT];
    #pragma unroll
    for (int mt = 0; mt < MT; ++mt) { acc0[mt] = vzero8(); acc1[mt] = vzero8(); }

    #pragma unroll
    for (int ks = 0; ks < KS; ++ks) {
        const int kb = ks * 32 + ((lane < 16) ? 0 : 16);
        BF16x16 bu0, bu1;
        if constexpr (XL == 0) {
            loadB_planes((const float*)Xb, ncol0, kb, bu0);
            loadB_planes((const float*)Xb, ncol1, kb, bu1);
            if (ks + 1 < KS) {   // ncol0/ncol1 share a 128B line: one prefetch each row-pair
                #pragma unroll
                for (int d = 0; d < 8; ++d)
                    __builtin_prefetch((const float*)Xb + (size_t)(kb + 32 + 2 * d) * HW + ncol0, 0, 3);
            }
        } else {
            loadB_tok<Kdim>((const unsigned short*)Xb, ncol0, kb, bu0);
            loadB_tok<Kdim>((const unsigned short*)Xb, ncol1, kb, bu1);
        }
        #pragma unroll
        for (int mt = 0; mt < MT; ++mt) {
            BF16x16 au;
            const unsigned int* p = &wl[((ks * MT + mt) * 32 + lane) * 8];
            #pragma unroll
            for (int d = 0; d < 8; ++d) au.u[d] = p[d];
            acc0[mt] = __builtin_amdgcn_wmma_f32_16x16x32_bf16(
                false, au.v, false, bu0.v, (short)0, acc0[mt], false, false);
            acc1[mt] = __builtin_amdgcn_wmma_f32_16x16x32_bf16(
                false, au.v, false, bu1.v, (short)0, acc1[mt], false, false);
        }
    }

    // Epilogue. C/D layout: lane<16: M = mt*16+r; lane>=16: M = mt*16+8+r.
    #pragma unroll
    for (int mt = 0; mt < MT; ++mt) {
        if constexpr (EPI == 0) {
            uint4 q0, q1;
            q0.x = packf(acc0[mt][0], acc0[mt][1]);
            q0.y = packf(acc0[mt][2], acc0[mt][3]);
            q0.z = packf(acc0[mt][4], acc0[mt][5]);
            q0.w = packf(acc0[mt][6], acc0[mt][7]);
            q1.x = packf(acc1[mt][0], acc1[mt][1]);
            q1.y = packf(acc1[mt][2], acc1[mt][3]);
            q1.z = packf(acc1[mt][4], acc1[mt][5]);
            q1.w = packf(acc1[mt][6], acc1[mt][7]);
            const int moff = mt * 16 + ((lane < 16) ? 0 : 8);
            *(uint4*)((unsigned short*)Y + ((size_t)b * HW + ncol0) * Mdim + moff) = q0;
            *(uint4*)((unsigned short*)Y + ((size_t)b * HW + ncol1) * Mdim + moff) = q1;
        } else {
            #pragma unroll
            for (int r = 0; r < 8; ++r) {
                int row = mt * 16 + r + ((lane < 16) ? 0 : 8);
                float inv = gamma[row] * rsqrtf(1.f + eps);
                size_t o0 = ((size_t)b * Mdim + row) * HW + ncol0;
                size_t o1 = o0 + 16;
                float y0 = acc0[mt][r] * inv + beta[row];
                float y1 = acc1[mt][r] * inv + beta[row];
                if constexpr (EPI == 1) { y0 += res[o0]; y1 += res[o1]; }
                else                    { y0 = silu(y0); y1 = silu(y1); }
                ((float*)Y)[o0] = y0;
                ((float*)Y)[o1] = y1;
            }
        }
    }
}

// ---------------------------------------------------------------------------
// 7x7 window attention. One wave per window.
// qkvT = bf16 token-major [b][HW][192] (q:0-63, k:64-127, v:128-191).
// outT = bf16 token-major [b][HW][64].
// ---------------------------------------------------------------------------
__global__ __launch_bounds__(32)
void attn_win(const unsigned short* __restrict__ qkvT, unsigned short* __restrict__ outT)
{
    __shared__ unsigned short Qt[64 * 64];   // [token][ch]; reused as P[i][j]
    __shared__ unsigned short Kt[64 * 64];   // [token][ch]
    __shared__ unsigned short Vtok[64 * 64]; // [token][ch]
    __shared__ float          S[64 * 64];    // scores [i][j]

    const int lane = threadIdx.x;
    const int win  = blockIdx.x;
    const int b    = blockIdx.y;
    const int wy = win >> 5, wx = win & 31;
    const int sbase = wy * 7 * IMG + wx * 7;

    // Coalesced token-major load: 96 dwords (192 bf16) per token.
    {
        const unsigned int* src = (const unsigned int*)qkvT;
        unsigned int* Qd = (unsigned int*)Qt;
        unsigned int* Kd = (unsigned int*)Kt;
        unsigned int* Vd = (unsigned int*)Vtok;
        for (int i = lane; i < 49 * 96; i += 32) {
            int t = i / 96, d = i % 96;
            int n = sbase + (t / 7) * IMG + (t % 7);
            unsigned int v = src[((size_t)b * HW + n) * 96 + d];
            if (d < 32)      Qd[t * 32 + d]      = v;
            else if (d < 64) Kd[t * 32 + d - 32] = v;
            else             Vd[t * 32 + d - 64] = v;
        }
        for (int i = lane; i < 15 * 96; i += 32) {  // zero pad tokens 49..63
            int t = 49 + i / 96, d = i % 96;
            if (d < 32)      Qd[t * 32 + d]      = 0;
            else if (d < 64) Kd[t * 32 + d - 32] = 0;
            else             Vd[t * 32 + d - 64] = 0;
        }
    }
    __syncthreads();

    const int mrow = (lane < 16) ? lane : (lane - 16);
    const int ncl  = lane & 15;

    // S = Q^T K  (M=tokens, N=tokens, K=64 channels)
    #pragma unroll
    for (int mt = 0; mt < 4; ++mt) {
        v8f acc[4];
        #pragma unroll
        for (int nt = 0; nt < 4; ++nt) acc[nt] = vzero8();
        #pragma unroll
        for (int ks = 0; ks < 2; ++ks) {
            const int kb = ks * 32;
            BF16x16 au;
            {
                int m = mt * 16 + mrow;
                int ko = kb + ((lane < 16) ? 0 : 16);
                const unsigned int* p0 = (const unsigned int*)&Qt[m * 64 + ko];
                const unsigned int* p1 = (const unsigned int*)&Qt[m * 64 + ko + 8];
                #pragma unroll
                for (int d = 0; d < 4; ++d) { au.u[d] = p0[d]; au.u[d + 4] = p1[d]; }
            }
            #pragma unroll
            for (int nt = 0; nt < 4; ++nt) {
                BF16x16 bu;
                int n  = nt * 16 + ncl;
                int ko = kb + ((lane < 16) ? 0 : 16);
                const unsigned int* p = (const unsigned int*)&Kt[n * 64 + ko];
                #pragma unroll
                for (int d = 0; d < 8; ++d) bu.u[d] = p[d];
                acc[nt] = __builtin_amdgcn_wmma_f32_16x16x32_bf16(
                    false, au.v, false, bu.v, (short)0, acc[nt], false, false);
            }
        }
        #pragma unroll
        for (int nt = 0; nt < 4; ++nt)
            #pragma unroll
            for (int r = 0; r < 8; ++r) {
                int i = mt * 16 + r + ((lane < 16) ? 0 : 8);
                int j = nt * 16 + ncl;
                S[i * 64 + j] = acc[nt][r] * 0.125f;  // SCALE = 64^-0.5
            }
    }
    __syncthreads();

    // softmax rows (mask j>=49), write P bf16 over Qt as [i][j].
    for (int rr = 0; rr < 2; ++rr) {
        int i = lane + rr * 32;
        if (i < 49) {
            float mx = -1e30f;
            for (int j = 0; j < 49; ++j) mx = fmaxf(mx, S[i * 64 + j]);
            float sum = 0.f;
            for (int j = 0; j < 49; ++j) sum += __expf(S[i * 64 + j] - mx);
            float rs = 1.f / sum;
            for (int j = 0; j < 64; ++j)
                Qt[i * 64 + j] = (j < 49) ? f2bf(__expf(S[i * 64 + j] - mx) * rs) : 0;
        } else {
            for (int j = 0; j < 64; ++j) Qt[i * 64 + j] = 0;
        }
    }
    __syncthreads();

    // O = V^T P^T : D[c][i] = sum_j V[j][c] P[i][j]; A strided from Vtok.
    #pragma unroll
    for (int nt = 0; nt < 4; ++nt) {
        v8f acc[4];
        #pragma unroll
        for (int mt = 0; mt < 4; ++mt) acc[mt] = vzero8();
        #pragma unroll
        for (int ks = 0; ks < 2; ++ks) {
            const int kb = ks * 32;
            BF16x16 bu;
            {
                int n  = nt * 16 + ncl;
                int ko = kb + ((lane < 16) ? 0 : 16);
                const unsigned int* p = (const unsigned int*)&Qt[n * 64 + ko];
                #pragma unroll
                for (int d = 0; d < 8; ++d) bu.u[d] = p[d];
            }
            #pragma unroll
            for (int mt = 0; mt < 4; ++mt) {
                BF16x16 au;
                int m  = mt * 16 + mrow;                 // channel
                int kA = kb + ((lane < 16) ? 0 : 16);    // token index base
                #pragma unroll
                for (int d = 0; d < 4; ++d) {
                    au.u[d]     = (unsigned int)Vtok[(kA + 2 * d) * 64 + m] |
                                  ((unsigned int)Vtok[(kA + 2 * d + 1) * 64 + m] << 16);
                    au.u[d + 4] = (unsigned int)Vtok[(kA + 8 + 2 * d) * 64 + m] |
                                  ((unsigned int)Vtok[(kA + 9 + 2 * d) * 64 + m] << 16);
                }
                acc[mt] = __builtin_amdgcn_wmma_f32_16x16x32_bf16(
                    false, au.v, false, bu.v, (short)0, acc[mt], false, false);
            }
        }
        // token-major vector store: 8 consecutive channels per lane.
        int i = nt * 16 + ncl;
        if (i < 49) {
            int n = sbase + (i / 7) * IMG + (i % 7);
            #pragma unroll
            for (int mt = 0; mt < 4; ++mt) {
                uint4 q;
                q.x = packf(acc[mt][0], acc[mt][1]);
                q.y = packf(acc[mt][2], acc[mt][3]);
                q.z = packf(acc[mt][4], acc[mt][5]);
                q.w = packf(acc[mt][6], acc[mt][7]);
                unsigned short* dst = outT +
                    ((size_t)b * HW + n) * 64 + mt * 16 + ((lane < 16) ? 0 : 8);
                *(uint4*)dst = q;
            }
        }
    }
}

// ---------------------------------------------------------------------------
// Fused: t = softmax_ch(conv3x3(x, wt, pad=1))  (8 ch), then
// y = silu(bn_{1e-3}(wp @ t)) + x.  16x16 spatial tile per 256-thread block.
// ---------------------------------------------------------------------------
__global__ __launch_bounds__(256)
void tmpl_block(const float* __restrict__ x, const float* __restrict__ wt,
                const float* __restrict__ wp, const float* __restrict__ gp,
                const float* __restrict__ bp, float* __restrict__ y)
{
    __shared__ float wtl[8 * NCH * 9];
    __shared__ float wpl[NCH * 8];
    __shared__ float tile[8][18][18];

    const int tid = threadIdx.x;
    const int b = blockIdx.z, th = blockIdx.y, tw = blockIdx.x;
    for (int i = tid; i < 8 * NCH * 9; i += 256) wtl[i] = wt[i];
    for (int i = tid; i < NCH * 8; i += 256)     wpl[i] = wp[i];

    const int ty = tid >> 4, tx = tid & 15;
    const int h = th * 16 + ty, w = tw * 16 + tx;
    const float* xb = x + (size_t)b * NCH * HW;

    float t[8] = {0.f,0.f,0.f,0.f,0.f,0.f,0.f,0.f};
    for (int cc = 0; cc < NCH; cc += 8) {
        __syncthreads();
        for (int i = tid; i < 8 * 18 * 18; i += 256) {
            int c = i / 324, rem = i % 324;
            int yy = rem / 18, xx = rem % 18;
            int hh = th * 16 + yy - 1, ww = tw * 16 + xx - 1;
            float v = 0.f;
            if (hh >= 0 && hh < IMG && ww >= 0 && ww < IMG)
                v = xb[(size_t)(cc + c) * HW + hh * IMG + ww];
            tile[c][yy][xx] = v;
        }
        __syncthreads();
        #pragma unroll
        for (int c = 0; c < 8; ++c)
            #pragma unroll
            for (int ky = 0; ky < 3; ++ky)
                #pragma unroll
                for (int kx = 0; kx < 3; ++kx) {
                    float xv = tile[c][ty + ky][tx + kx];
                    #pragma unroll
                    for (int o = 0; o < 8; ++o)
                        t[o] = fmaf(wtl[((o * NCH + cc + c) * 3 + ky) * 3 + kx], xv, t[o]);
                }
    }
    float mx = t[0];
    #pragma unroll
    for (int o = 1; o < 8; ++o) mx = fmaxf(mx, t[o]);
    float sum = 0.f;
    #pragma unroll
    for (int o = 0; o < 8; ++o) { t[o] = __expf(t[o] - mx); sum += t[o]; }
    float rs = 1.f / sum;
    #pragma unroll
    for (int o = 0; o < 8; ++o) t[o] *= rs;

    const size_t pix = (size_t)b * NCH * HW + (size_t)h * IMG + w;
    for (int c = 0; c < NCH; ++c) {
        float a = 0.f;
        #pragma unroll
        for (int j = 0; j < 8; ++j) a = fmaf(wpl[c * 8 + j], t[j], a);
        float inv = gp[c] * rsqrtf(1.f + 1e-3f);
        float v = silu(a * inv + bp[c]);
        y[pix + (size_t)c * HW] = v + x[pix + (size_t)c * HW];
    }
}

// ---------------------------------------------------------------------------
// Depthwise 3x3 (pad 1) + BN(1e-5) + SiLU.
// ---------------------------------------------------------------------------
__global__ __launch_bounds__(256)
void dwconv(const float* __restrict__ x, const float* __restrict__ wd,
            const float* __restrict__ g, const float* __restrict__ bb,
            float* __restrict__ y)
{
    size_t idx = (size_t)blockIdx.x * 256 + threadIdx.x;
    if (idx >= (size_t)8 * NCH * HW) return;
    int w = (int)(idx % IMG);
    size_t r = idx / IMG;
    int h = (int)(r % IMG); r /= IMG;
    int c = (int)(r % NCH);
    const float* xp = x + (idx - (size_t)(h * IMG + w));
    float acc = 0.f;
    #pragma unroll
    for (int ky = 0; ky < 3; ++ky) {
        int hh = h + ky - 1;
        if (hh < 0 || hh >= IMG) continue;
        #pragma unroll
        for (int kx = 0; kx < 3; ++kx) {
            int ww = w + kx - 1;
            if (ww < 0 || ww >= IMG) continue;
            acc = fmaf(wd[c * 9 + ky * 3 + kx], xp[hh * IMG + ww], acc);
        }
    }
    float inv = g[c] * rsqrtf(1.f + 1e-5f);
    y[idx] = silu(acc * inv + bb[c]);
}

__global__ void concat_qkv(const float* __restrict__ q, const float* __restrict__ k,
                           const float* __restrict__ v, float* __restrict__ o)
{
    int i = blockIdx.x * 256 + threadIdx.x;
    if (i < 8192)       o[i] = q[i];
    else if (i < 16384) o[i] = k[i - 8192];
    else if (i < 24576) o[i] = v[i - 16384];
}

extern "C" void kernel_launch(void* const* d_in, const int* in_sizes, int n_in,
                              void* d_out, int out_size, void* d_ws, size_t ws_size,
                              hipStream_t stream)
{
    const float* x   = (const float*)d_in[0];
    const float* wq  = (const float*)d_in[1];
    const float* wk  = (const float*)d_in[2];
    const float* wv  = (const float*)d_in[3];
    const float* wo  = (const float*)d_in[4];
    const float* go  = (const float*)d_in[5];
    const float* bo  = (const float*)d_in[6];
    const float* wt  = (const float*)d_in[7];
    const float* wp  = (const float*)d_in[8];
    const float* gp  = (const float*)d_in[9];
    const float* bp  = (const float*)d_in[10];
    const float* dw1 = (const float*)d_in[11];
    const float* g1  = (const float*)d_in[12];
    const float* b1  = (const float*)d_in[13];
    const float* pw1 = (const float*)d_in[14];
    const float* g2  = (const float*)d_in[15];
    const float* b2  = (const float*)d_in[16];
    const float* dw2 = (const float*)d_in[17];
    const float* g3  = (const float*)d_in[18];
    const float* b3  = (const float*)d_in[19];
    const float* pw2 = (const float*)d_in[20];
    const float* g4  = (const float*)d_in[21];
    const float* b4  = (const float*)d_in[22];

    char* ws = (char*)d_ws;
    size_t off = 0;
    float* Wqkv = (float*)(ws + off);           off += 192 * 128 * sizeof(float);
    off = (off + 255) & ~(size_t)255;
    unsigned short* qkvT = (unsigned short*)(ws + off); off += (size_t)8 * HW * 192 * 2;
    unsigned short* attnT = (unsigned short*)(ws + off); off += (size_t)8 * HW * 64 * 2;
    float* xA = (float*)(ws + off);             off += (size_t)8 * NCH * HW * 4;
    float* xB = (float*)(ws + off);

    const dim3 gemm_grid(HW / 128, 8);          // 32 cols/wave * 4 waves/block
    const size_t total = (size_t)8 * NCH * HW;
    const int dwblocks = (int)((total + 255) / 256);

    // 1) concat qkv weights into [192][128]
    concat_qkv<<<96, 256, 0, stream>>>(wq, wk, wv, Wqkv);
    // 2) qkv projection -> bf16 token-major [b][HW][192]
    gemm_wmma<192, 128, float, 0, 0><<<gemm_grid, 128, 0, stream>>>(
        Wqkv, x, qkvT, nullptr, nullptr, 0.f, nullptr);
    // 3) window attention -> bf16 token-major [b][HW][64]
    attn_win<<<dim3(1024, 8), 32, 0, stream>>>(qkvT, attnT);
    // 4) out projection + BN(1e-5) + residual -> xA (fp32 planes)
    gemm_wmma<128, 64, unsigned short, 1, 1><<<gemm_grid, 128, 0, stream>>>(
        wo, attnT, xA, go, bo, 1e-5f, x);
    // 5) template conv + softmax + 1x1 + BN(1e-3) + SiLU + residual -> xB
    tmpl_block<<<dim3(14, 14, 8), 256, 0, stream>>>(xA, wt, wp, gp, bp, xB);
    // 6) dw1 + BN + SiLU -> xA
    dwconv<<<dwblocks, 256, 0, stream>>>(xB, dw1, g1, b1, xA);
    // 7) pw1 + BN + SiLU -> xB
    gemm_wmma<128, 128, float, 2, 0><<<gemm_grid, 128, 0, stream>>>(
        pw1, xA, xB, g2, b2, 1e-5f, nullptr);
    // 8) dw2 + BN + SiLU -> xA
    dwconv<<<dwblocks, 256, 0, stream>>>(xB, dw2, g3, b3, xA);
    // 9) pw2 + BN + SiLU -> d_out
    gemm_wmma<128, 128, float, 2, 0><<<gemm_grid, 128, 0, stream>>>(
        pw2, xA, (float*)d_out, g4, b4, 1e-5f, nullptr);
}
